// ModulatedConv2d_74302934221259
// MI455X (gfx1250) — matrix operand order repaired and checked by reading, not compiled, not run on hardware
//
#include <hip/hip_runtime.h>
#include <hip/hip_bf16.h>
#include <stddef.h>

// ---------------------------------------------------------------------------
// ModulatedConv2d (StyleGAN2-style) for MI455X / gfx1250.
// Grouped 3x3 conv as 16 implicit GEMMs: M=512(out-ch) x N=4096(pixels) x
// K=4608(512 in-ch * 9 taps), via v_wmma_f32_16x16x32_bf16 (f32 accumulate).
// Workgroup tile 128(M) x 256(N); 8 waves, each owning a 64x64 sub-tile
// (4x4 WMMA accumulators). Double-buffered LDS, one barrier per K-step.
// B operand pre-converted to bf16 NHWC so the hot loop does only b128 loads.
// ---------------------------------------------------------------------------

typedef __attribute__((ext_vector_type(16))) __bf16        v16bf;
typedef __attribute__((ext_vector_type(8)))  float         v8f;
typedef __attribute__((ext_vector_type(8)))  unsigned int  v8u;

#define IN_CH   512
#define OUT_CH  512
#define KTOT    4608             // 9 * 512, k = (kh*3+kw)*512 + i
#define KSTEP   32
#define KITERS  (KTOT / KSTEP)   // 144
#define HW      4096             // 64*64 pixels
#define CTOT    2048             // 4 groups * 512 channels
#define MTILE   128
#define NTILE   256

__device__ __forceinline__ unsigned short f2bf(float f) {
    unsigned int u = __builtin_bit_cast(unsigned int, f);
    u += 0x7fffu + ((u >> 16) & 1u);     // round-to-nearest-even
    return (unsigned short)(u >> 16);
}
__device__ __forceinline__ unsigned int pack_bf16(float a, float b) {
    return (unsigned int)f2bf(a) | ((unsigned int)f2bf(b) << 16);
}

// s[g][i] = mod_b[i] + sum_j style[g][j] * mod_w[i][j]      (4 x 512)
__global__ __launch_bounds__(256) void style_mod_kernel(
    const float* __restrict__ style, const float* __restrict__ mod_w,
    const float* __restrict__ mod_b, float* __restrict__ s)
{
    int idx = blockIdx.x * 256 + threadIdx.x;   // 0..2047
    int g = idx >> 9, i = idx & 511;
    const float* st = style + g * 512;
    const float* mw = mod_w + i * 512;
    float acc = mod_b[i];
    #pragma unroll 8
    for (int j = 0; j < 512; ++j) acc += st[j] * mw[j];
    s[idx] = acc;
}

// wA[g][o][(kh*3+kw)*512 + i] = bf16( weight[o][i][kh][kw] * s[g][i] * demod[o] )
__global__ __launch_bounds__(256) void mod_weight_kernel(
    const float* __restrict__ weight, const float* __restrict__ demod,
    const float* __restrict__ s, unsigned short* __restrict__ wA)
{
    int k = blockIdx.x * 256 + threadIdx.x;     // 0..4607
    int o = blockIdx.y;                          // 0..511
    int g = blockIdx.z;                          // 0..3
    int r = k >> 9;          // tap index kh*3+kw
    int i = k & 511;         // input channel
    float v = weight[(size_t)o * KTOT + i * 9 + r] * s[g * 512 + i] * demod[o];
    wA[((size_t)g * OUT_CH + o) * KTOT + k] = f2bf(v);
}

// x (batch, 2048, 64, 64) f32  ->  xb (batch, 4096 pix, 2048 ch) bf16 (NHWC)
__global__ __launch_bounds__(256) void x_nhwc_bf16_kernel(
    const float* __restrict__ x, unsigned short* __restrict__ xb)
{
    __shared__ unsigned short tile[32][33];
    const int t     = threadIdx.x;
    const int p0    = blockIdx.x * 32;
    const int c0    = blockIdx.y * 32;
    const int batch = blockIdx.z;
    {   // read 32c x 32p, coalesced along p
        const int pl = t & 31, cl = t >> 5;
        const float* xp = x + ((size_t)batch * CTOT + c0 + cl) * HW + p0 + pl;
        #pragma unroll
        for (int i = 0; i < 4; ++i)
            tile[cl + 8 * i][pl] = f2bf(xp[(size_t)(8 * i) * HW]);
    }
    __syncthreads();
    {   // write 32p x 32c, coalesced along c
        const int cl = t & 31, pl = t >> 5;
        unsigned short* op = xb + ((size_t)batch * HW + p0 + pl) * CTOT + c0 + cl;
        #pragma unroll
        for (int i = 0; i < 4; ++i)
            op[(size_t)(8 * i) * CTOT] = tile[cl][pl + 8 * i];
    }
}

template <bool NHWC>
__global__ __launch_bounds__(256) void conv_wmma_kernel(
    const float* __restrict__ x, const unsigned short* __restrict__ xb,
    const unsigned short* __restrict__ wA, float* __restrict__ out)
{
    // Packed-dword LDS staging in WMMA fragment layout. Row stride 20 dwords:
    // every 4-dword group is 16B-aligned and 20*row mod 64 walks all 16
    // distinct 4-bank blocks -> conflict-friendly b128 LDS ops.
    __shared__ __align__(16) unsigned int Ap[2][MTILE][20];   // [m][kpair]
    __shared__ __align__(16) unsigned int Bp[2][NTILE][20];   // [n][kpair]

    const int tid    = threadIdx.x;
    const int lane   = tid & 31;
    const int l15    = lane & 15;
    const int khalf  = lane >> 4;
    const int wid    = tid >> 5;                // 8 waves: 2 (M) x 4 (N)
    const int wave_m = (wid >> 2) * 64;
    const int wave_n = (wid & 3) * 64;

    const int n0    = blockIdx.x * NTILE;       // pixel tile (16 tiles)
    const int m0    = blockIdx.y * MTILE;       // out-channel tile (4 tiles)
    const int bz    = blockIdx.z;               // 16 = batch*4 + group
    const int batch = bz >> 2;
    const int g     = bz & 3;

    const unsigned short* wAg = wA + (size_t)g * OUT_CH * KTOT + (size_t)m0 * KTOT;
    const float*          xg  = x  + ((size_t)batch * CTOT + g * IN_CH) * HW;
    const unsigned short* xbg = xb + (size_t)batch * HW * CTOT + g * IN_CH;

    v8f acc[4][4] = {};
    unsigned int ra[8], rb[16];

    // A tile: 128 m x 16 kpair dwords. Thread t: fixed kpair = t&15, rows t>>4 + 16*i.
    auto loadA = [&](int k0, unsigned int* r) {
        const int kk = tid & 15;
        const int mb = tid >> 4;
        const unsigned short* p = wAg + (size_t)mb * KTOT + k0 + kk * 2;
        #pragma unroll
        for (int i = 0; i < 8; ++i)
            r[i] = *(const unsigned int*)(p + (size_t)(16 * i) * KTOT);
    };
    // B tile: 256 n x 16 kpair dwords. Thread t: fixed pixel n = t, kpairs 0..15.
    // One K-step lies entirely inside one conv tap -> constant (dy,dx).
    auto loadB = [&](int k0, unsigned int* r) {
        const int tap = k0 >> 9;
        const int t3  = tap / 3;
        const int dy  = t3 - 1, dx = (tap - 3 * t3) - 1;
        const int ib  = k0 & 511;               // base input channel
        const int p   = n0 + tid;
        const int yy  = (p >> 6) + dy;
        const int xs  = (p & 63) + dx;
        const bool ok = ((unsigned)yy < 64u) && ((unsigned)xs < 64u);
        if (NHWC) {
            if (ok) {   // 32 contiguous bf16 channels at one pixel: 4 x b128
                const uint4* bp =
                    (const uint4*)(xbg + (size_t)(yy * 64 + xs) * CTOT + ib);
                #pragma unroll
                for (int i = 0; i < 4; ++i) {
                    uint4 v = bp[i];
                    r[4*i+0] = v.x; r[4*i+1] = v.y; r[4*i+2] = v.z; r[4*i+3] = v.w;
                }
            } else {
                #pragma unroll
                for (int i = 0; i < 16; ++i) r[i] = 0u;
            }
        } else {        // fallback: gather f32 NCHW + convert
            const float* px = xg + (size_t)ib * HW + yy * 64 + xs;
            #pragma unroll
            for (int i = 0; i < 16; ++i) {
                unsigned int val = 0u;
                if (ok) val = pack_bf16(px[(size_t)(2 * i) * HW],
                                        px[(size_t)(2 * i + 1) * HW]);
                r[i] = val;
            }
        }
    };
    auto stage = [&](int buf) {
        const int kk = tid & 15, mb = tid >> 4;
        #pragma unroll
        for (int i = 0; i < 8; ++i) Ap[buf][mb + 16 * i][kk] = ra[i];
        uint4* bdst = (uint4*)&Bp[buf][tid][0];       // 4 x ds_store_b128
        #pragma unroll
        for (int i = 0; i < 4; ++i)
            bdst[i] = make_uint4(rb[4*i], rb[4*i+1], rb[4*i+2], rb[4*i+3]);
    };

    loadA(0, ra); loadB(0, rb);
    stage(0);
    __syncthreads();

    for (int t = 0; t < KITERS; ++t) {
        const int buf = t & 1;
        if (t + 1 < KITERS) { loadA((t + 1) * KSTEP, ra); loadB((t + 1) * KSTEP, rb); }

        // B fragments: VGPR 0-3 = kpairs [khalf*4..+3], VGPR 4-7 = [8+khalf*4..+3]: 2 x b128
        v16bf bfrag[4];
        #pragma unroll
        for (int nt = 0; nt < 4; ++nt) {
            const int col = wave_n + nt * 16 + l15;
            uint4 lo = *(const uint4*)&Bp[buf][col][khalf * 4];
            uint4 hi = *(const uint4*)&Bp[buf][col][8 + khalf * 4];
            v8u u = {lo.x, lo.y, lo.z, lo.w, hi.x, hi.y, hi.z, hi.w};
            bfrag[nt] = __builtin_bit_cast(v16bf, u);
        }
        // A fragments: VGPR 0-3 = kpairs [khalf*4..+3], VGPR 4-7 = [8+khalf*4..+3]
        #pragma unroll
        for (int mt = 0; mt < 4; ++mt) {
            const int row = wave_m + mt * 16 + l15;
            uint4 lo = *(const uint4*)&Ap[buf][row][khalf * 4];
            uint4 hi = *(const uint4*)&Ap[buf][row][8 + khalf * 4];
            v8u u = {lo.x, lo.y, lo.z, lo.w, hi.x, hi.y, hi.z, hi.w};
            v16bf afrag = __builtin_bit_cast(v16bf, u);
            #pragma unroll
            for (int nt = 0; nt < 4; ++nt)
                acc[mt][nt] = __builtin_amdgcn_wmma_f32_16x16x32_bf16(
                    false, afrag, false, bfrag[nt],
                    (short)0, acc[mt][nt], false, false);
        }

        if (t + 1 < KITERS) stage(buf ^ 1);
        __syncthreads();
    }

    // C layout: lanes 0-15 -> M=v, N=lane; lanes 16-31 -> M=8+v, N=lane-16
    float* outb = out + ((size_t)batch * CTOT + g * OUT_CH + m0) * HW;
    #pragma unroll
    for (int mt = 0; mt < 4; ++mt)
        #pragma unroll
        for (int nt = 0; nt < 4; ++nt)
            #pragma unroll
            for (int v = 0; v < 8; ++v) {
                int o = wave_m + mt * 16 + khalf * 8 + v;
                int p = n0 + wave_n + nt * 16 + l15;
                outb[(size_t)o * HW + p] = acc[mt][nt][v];
            }
}

extern "C" void kernel_launch(void* const* d_in, const int* in_sizes, int n_in,
                              void* d_out, int out_size, void* d_ws, size_t ws_size,
                              hipStream_t stream) {
    const float* x      = (const float*)d_in[0];   // (4, 2048, 64, 64)
    const float* style  = (const float*)d_in[1];   // (4, 512)
    const float* weight = (const float*)d_in[2];   // (1, 512, 512, 3, 3)
    const float* demod  = (const float*)d_in[3];   // (1, 512, 1, 1)
    const float* mod_w  = (const float*)d_in[4];   // (512, 512)
    const float* mod_b  = (const float*)d_in[5];   // (512,)
    float* out = (float*)d_out;                    // (4, 2048, 64, 64)

    const size_t WA_OFF   = 8192;
    const size_t WA_BYTES = (size_t)4 * OUT_CH * KTOT * 2;      // 18,874,368
    const size_t XB_OFF   = WA_OFF + WA_BYTES;
    const size_t XB_BYTES = (size_t)4 * HW * CTOT * 2;          // 67,108,864

    float*          s_buf = (float*)d_ws;
    unsigned short* wA    = (unsigned short*)((char*)d_ws + WA_OFF);
    unsigned short* xb    = (unsigned short*)((char*)d_ws + XB_OFF);

    style_mod_kernel<<<8, 256, 0, stream>>>(style, mod_w, mod_b, s_buf);

    dim3 gw(KTOT / 256, OUT_CH, 4);
    mod_weight_kernel<<<gw, 256, 0, stream>>>(weight, demod, s_buf, wA);

    dim3 gc(HW / NTILE, OUT_CH / MTILE, 16);   // (16, 4, 16)
    if (ws_size >= XB_OFF + XB_BYTES) {
        dim3 gt(HW / 32, CTOT / 32, 4);        // (128, 64, 4)
        x_nhwc_bf16_kernel<<<gt, 256, 0, stream>>>(x, xb);
        conv_wmma_kernel<true><<<gc, 256, 0, stream>>>(x, xb, wA, out);
    } else {
        conv_wmma_kernel<false><<<gc, 256, 0, stream>>>(x, xb, wA, out);
    }
}
